// PointNetFeaturePropagation_90632399880316
// MI455X (gfx1250) — matrix operand order repaired and checked by reading, not compiled, not run on hardware
//
#include <hip/hip_runtime.h>
#include <math.h>

// ---------------------------------------------------------------------------
// PointNet Feature Propagation for MI455X (gfx1250), wave32 + WMMA bf16.
//
// Pipeline (all on `stream`):
//   1. zero stats
//   2. convert conv weights f32 -> bf16
//   3. nearest-neighbor argmin (xyz2 tile in LDS)
//   4. gather/concat -> Xcat bf16 [P][384]  (point-major, K contiguous)
//   5. GEMM0 (WMMA bf16 -> f32 acc) -> Y0 bf16 [256][P]
//   6. per-channel sum/sumsq -> finalize BN scale/shift
//   7. BN+ReLU + transpose -> X1 bf16 [P][256]
//   8. GEMM1 -> Y1 f32 [256][P] (aliases dead Xcat region)
//   9. stats + finalize
//  10. BN+ReLU -> d_out f32 [B][256][N]
// ---------------------------------------------------------------------------

typedef __attribute__((ext_vector_type(16))) __bf16 v16bf;
typedef __attribute__((ext_vector_type(8)))  __bf16 v8bf;
typedef __attribute__((ext_vector_type(8)))  float  v8f;

#define PB    8
#define NB    8192
#define MB    2048
#define PTS   65536        // PB * NB
#define CIN0  384
#define CMID  256
#define BN_EPS 1e-5f

__device__ __forceinline__ v8f wmma_bf16(v16bf a, v16bf b, v8f c) {
  // D = A(16x32) * B(32x16) + C, f32 accumulate
  return __builtin_amdgcn_wmma_f32_16x16x32_bf16(
      /*neg_a=*/false, a, /*neg_b=*/false, b,
      /*c_mod=*/(short)0, c, /*reuse_a=*/false, /*reuse_b=*/false);
}

// ---------------------------------------------------------------------------
// 0. zero the stat accumulators (harness poisons ws only once; atomics below
//    require fresh zeros every call)
// ---------------------------------------------------------------------------
__global__ void zero_stats_kernel(float* __restrict__ s) {
  s[blockIdx.x * 256 + threadIdx.x] = 0.0f;
}

// ---------------------------------------------------------------------------
// 1. weights f32 -> bf16
// ---------------------------------------------------------------------------
__global__ void cvt_w_kernel(const float* __restrict__ w0, const float* __restrict__ w1,
                             __bf16* __restrict__ w0b, __bf16* __restrict__ w1b) {
  int t = blockIdx.x * 256 + threadIdx.x;
  if (t < CMID * CIN0) w0b[t] = (__bf16)w0[t];
  if (t < CMID * CMID) w1b[t] = (__bf16)w1[t];
}

// ---------------------------------------------------------------------------
// 2. nearest neighbor: idx[b*NB+n] = argmin_m ||xyz1[b,n]-xyz2[b,m]||^2
//    xyz2[b] (2048*3 f32 = 24KB) staged in LDS; first-hit tie break matches argmin.
// ---------------------------------------------------------------------------
__global__ __launch_bounds__(256)
void nn_kernel(const float* __restrict__ xyz1, const float* __restrict__ xyz2,
               int* __restrict__ idx) {
  __shared__ float sx[MB], sy[MB], sz[MB];
  const int b = blockIdx.y;
  for (int j = threadIdx.x; j < MB; j += 256) {
    const float* q = xyz2 + ((size_t)b * MB + j) * 3;
    sx[j] = q[0]; sy[j] = q[1]; sz[j] = q[2];
  }
  __syncthreads();
  const int n = blockIdx.x * 256 + threadIdx.x;
  const int p = b * NB + n;
  const float* q1 = xyz1 + (size_t)p * 3;
  const float x = q1[0], y = q1[1], z = q1[2];
  float best = 3.4e38f; int bi = 0;
  #pragma unroll 4
  for (int m = 0; m < MB; ++m) {
    float dx = x - sx[m], dy = y - sy[m], dz = z - sz[m];
    float d = dx * dx + dy * dy + dz * dz;
    if (d < best) { best = d; bi = m; }
  }
  idx[p] = bi;
}

// ---------------------------------------------------------------------------
// 3. build Xcat bf16 [P][384]: rows 0..127 = features1, 128..383 = gathered f2
// ---------------------------------------------------------------------------
__global__ __launch_bounds__(256)
void build_x_kernel(const float* __restrict__ f1, const float* __restrict__ f2,
                    const int* __restrict__ idx, __bf16* __restrict__ X) {
  size_t e = (size_t)blockIdx.x * 256 + threadIdx.x;   // over P*384, c fastest
  int p = (int)(e / CIN0);
  int c = (int)(e % CIN0);
  int b = p >> 13, n = p & (NB - 1);
  float v;
  if (c < 128) v = f1[((size_t)b * 128 + c) * NB + n];
  else         v = f2[((size_t)b * 256 + (c - 128)) * MB + idx[p]];
  X[e] = (__bf16)v;
}

// ---------------------------------------------------------------------------
// 4. WMMA GEMM: Y[C][P] = W[256][K] * X[P][K]^T   (bf16 in, f32 acc)
//    Wave tile: 128 channels x 32 points (8 M-tiles x 2 N-tiles, 16 wmma/K-step).
//    Block: 8 waves -> 256 channels x 128 points. EXEC is all-ones (no divergence).
// ---------------------------------------------------------------------------
template <int K, bool OUT_BF16>
__global__ __launch_bounds__(256)
void gemm_wmma_kernel(const __bf16* __restrict__ W, const __bf16* __restrict__ X,
                      void* __restrict__ Yv) {
  const int lane   = threadIdx.x & 31;
  const int wave   = threadIdx.x >> 5;
  const int h      = lane >> 4;       // half-wave selector
  const int l15    = lane & 15;
  const int cBase  = (wave & 1) * 128;            // channel group
  const int pBase  = blockIdx.x * 128 + (wave >> 1) * 32;  // point group

  v8f acc[8][2];
  #pragma unroll
  for (int mt = 0; mt < 8; ++mt)
    #pragma unroll
    for (int nt = 0; nt < 2; ++nt)
      acc[mt][nt] = (v8f)0.0f;

  for (int k = 0; k < K; k += 32) {
    // B fragments: lane holds point pBase+nt*16+l15, K = k+16h .. k+16h+15 (contiguous)
    v16bf bfrag[2];
    #pragma unroll
    for (int nt = 0; nt < 2; ++nt) {
      const __bf16* xp = X + (size_t)(pBase + nt * 16 + l15) * K + k + 16 * h;
      bfrag[nt] = *(const v16bf*)xp;
      if (k + 32 < K) __builtin_prefetch((const void*)(xp + 32), 0, 1);
    }
    #pragma unroll
    for (int mt = 0; mt < 8; ++mt) {
      // A fragment: row m = cBase+mt*16+l15; K = k+16h .. k+16h+15 (contiguous)
      const __bf16* wr = W + (size_t)(cBase + mt * 16 + l15) * K + k;
      union { v16bf v; v8bf half[2]; } au;
      au.half[0] = *(const v8bf*)(wr + 16 * h);
      au.half[1] = *(const v8bf*)(wr + 16 * h + 8);
      acc[mt][0] = wmma_bf16(au.v, bfrag[0], acc[mt][0]);
      acc[mt][1] = wmma_bf16(au.v, bfrag[1], acc[mt][1]);
    }
  }

  // D layout: VGPR r -> channel r+8h, lane l15 -> point
  #pragma unroll
  for (int mt = 0; mt < 8; ++mt)
    #pragma unroll
    for (int nt = 0; nt < 2; ++nt)
      #pragma unroll
      for (int r = 0; r < 8; ++r) {
        int ch = cBase + mt * 16 + r + 8 * h;
        int p  = pBase + nt * 16 + l15;
        if (OUT_BF16) ((__bf16*)Yv)[(size_t)ch * PTS + p] = (__bf16)acc[mt][nt][r];
        else          ((float*)Yv)[(size_t)ch * PTS + p]  = acc[mt][nt][r];
      }
}

// ---------------------------------------------------------------------------
// 5. per-channel sum / sumsq over P (rows contiguous in [C][P] layout)
// ---------------------------------------------------------------------------
template <typename T>
__global__ __launch_bounds__(256)
void stats_kernel(const T* __restrict__ Y, float* __restrict__ sum,
                  float* __restrict__ sumsq) {
  const int c = blockIdx.x;
  const size_t base = (size_t)c * PTS + (size_t)blockIdx.y * 8192;
  float s = 0.f, q = 0.f;
  for (int i = threadIdx.x; i < 8192; i += 256) {
    float v = (float)Y[base + i];
    s += v; q += v * v;
  }
  __shared__ float rs[256], rq[256];
  rs[threadIdx.x] = s; rq[threadIdx.x] = q;
  __syncthreads();
  for (int off = 128; off > 0; off >>= 1) {
    if (threadIdx.x < off) {
      rs[threadIdx.x] += rs[threadIdx.x + off];
      rq[threadIdx.x] += rq[threadIdx.x + off];
    }
    __syncthreads();
  }
  if (threadIdx.x == 0) { atomicAdd(&sum[c], rs[0]); atomicAdd(&sumsq[c], rq[0]); }
}

// ---------------------------------------------------------------------------
// 6. fold training-mode BN (biased var) into per-channel scale/shift
// ---------------------------------------------------------------------------
__global__ void finalize_stats_kernel(const float* __restrict__ sum,
                                      const float* __restrict__ sumsq,
                                      const float* __restrict__ gamma,
                                      const float* __restrict__ beta,
                                      float* __restrict__ scale,
                                      float* __restrict__ shift) {
  const int c = threadIdx.x;
  const float invP = 1.0f / (float)PTS;
  float mu  = sum[c] * invP;
  float var = fmaxf(sumsq[c] * invP - mu * mu, 0.0f);
  float sc  = gamma[c] * rsqrtf(var + BN_EPS);
  scale[c] = sc;
  shift[c] = beta[c] - mu * sc;
}

// ---------------------------------------------------------------------------
// 7. BN + ReLU + transpose: Y0 bf16 [C][P] -> X1 bf16 [P][C] (LDS 64x64 tile)
// ---------------------------------------------------------------------------
__global__ __launch_bounds__(256)
void bnrelu_transpose_kernel(const __bf16* __restrict__ Y,
                             const float* __restrict__ scale,
                             const float* __restrict__ shift,
                             __bf16* __restrict__ Xo) {
  __shared__ __bf16 tile[64][66];   // pad to dodge bank conflicts
  const int c0 = blockIdx.y * 64, p0 = blockIdx.x * 64;
  #pragma unroll
  for (int i = 0; i < 16; ++i) {
    int t = threadIdx.x + i * 256;
    int cl = t >> 6, pl = t & 63;
    int c = c0 + cl;
    float v = (float)Y[(size_t)c * PTS + p0 + pl];
    tile[cl][pl] = (__bf16)fmaxf(scale[c] * v + shift[c], 0.0f);
  }
  __syncthreads();
  #pragma unroll
  for (int i = 0; i < 16; ++i) {
    int t = threadIdx.x + i * 256;
    int pl = t >> 6, cl = t & 63;
    Xo[(size_t)(p0 + pl) * CMID + c0 + cl] = tile[cl][pl];
  }
}

// ---------------------------------------------------------------------------
// 8. final BN + ReLU, f32 output in reference layout [B][256][N]
// ---------------------------------------------------------------------------
__global__ __launch_bounds__(256)
void final_out_kernel(const float* __restrict__ Y, const float* __restrict__ scale,
                      const float* __restrict__ shift, float* __restrict__ out) {
  size_t e = (size_t)blockIdx.x * 256 + threadIdx.x;   // over B*256*N, n fastest
  int n = (int)(e & (NB - 1));
  int c = (int)((e >> 13) & 255);
  int b = (int)(e >> 21);
  float v = Y[(size_t)c * PTS + (size_t)b * NB + n];
  out[e] = fmaxf(scale[c] * v + shift[c], 0.0f);
}

// ---------------------------------------------------------------------------
// launch
// ---------------------------------------------------------------------------
extern "C" void kernel_launch(void* const* d_in, const int* in_sizes, int n_in,
                              void* d_out, int out_size, void* d_ws, size_t ws_size,
                              hipStream_t stream) {
  (void)in_sizes; (void)n_in; (void)out_size; (void)ws_size;
  const float* xyz1   = (const float*)d_in[0];
  const float* xyz2   = (const float*)d_in[1];
  const float* f1     = (const float*)d_in[2];
  const float* f2     = (const float*)d_in[3];
  const float* w0     = (const float*)d_in[4];
  const float* gamma0 = (const float*)d_in[5];
  const float* beta0  = (const float*)d_in[6];
  const float* w1     = (const float*)d_in[7];
  const float* gamma1 = (const float*)d_in[8];
  const float* beta1  = (const float*)d_in[9];
  float* out = (float*)d_out;

  // workspace layout (~135 MB); Y1 aliases Xcat (Xcat dead before GEMM1 runs)
  char* ws = (char*)d_ws;
  __bf16* Xcat = (__bf16*)(ws + 0);                 // 50.3 MB  [P][384]
  float*  Y1   = (float*) (ws + 0);                 // 67.1 MB  [256][P] (alias)
  __bf16* Y0   = (__bf16*)(ws + 67108864);          // 33.6 MB  [256][P]
  __bf16* X1   = (__bf16*)(ws + 100663296);         // 33.6 MB  [P][256]
  int*    idx  = (int*)   (ws + 134217728);         // 256 KB
  __bf16* W0b  = (__bf16*)(ws + 134479872);         // 192 KB
  __bf16* W1b  = (__bf16*)(ws + 134676480);         // 128 KB
  float*  st   = (float*) (ws + 134807552);         // stats block
  float *s0 = st, *ss0 = st + 256, *s1 = st + 512, *ss1 = st + 768;
  float *sc0 = st + 1024, *sh0 = st + 1280, *sc1 = st + 1536, *sh1 = st + 1792;

  zero_stats_kernel<<<4, 256, 0, stream>>>(st);
  cvt_w_kernel<<<(CMID * CIN0) / 256, 256, 0, stream>>>(w0, w1, W0b, W1b);

  nn_kernel<<<dim3(NB / 256, PB), 256, 0, stream>>>(xyz1, xyz2, idx);
  build_x_kernel<<<(unsigned)((size_t)PTS * CIN0 / 256), 256, 0, stream>>>(f1, f2, idx, Xcat);

  gemm_wmma_kernel<CIN0, true><<<PTS / 128, 256, 0, stream>>>(W0b, Xcat, (void*)Y0);
  stats_kernel<__bf16><<<dim3(CMID, PTS / 8192), 256, 0, stream>>>(Y0, s0, ss0);
  finalize_stats_kernel<<<1, 256, 0, stream>>>(s0, ss0, gamma0, beta0, sc0, sh0);
  bnrelu_transpose_kernel<<<dim3(PTS / 64, CMID / 64), 256, 0, stream>>>(Y0, sc0, sh0, X1);

  gemm_wmma_kernel<CMID, false><<<PTS / 128, 256, 0, stream>>>(W1b, X1, (void*)Y1);
  stats_kernel<float><<<dim3(CMID, PTS / 8192), 256, 0, stream>>>(Y1, s1, ss1);
  finalize_stats_kernel<<<1, 256, 0, stream>>>(s1, ss1, gamma1, beta1, sc1, sh1);

  final_out_kernel<<<(unsigned)((size_t)PB * CMID * NB / 256), 256, 0, stream>>>(Y1, sc1, sh1, out);
}
